// Simplified_PointNet_84430467104795
// MI455X (gfx1250) — matrix operand order repaired and checked by reading, not compiled, not run on hardware
//
#include <hip/hip_runtime.h>

// CDNA5 / gfx1250, wave32. Uses v_wmma_f32_16x16x32_f16 for every GEMM.
typedef __attribute__((ext_vector_type(16))) _Float16 v16h;
typedef __attribute__((ext_vector_type(8)))  _Float16 v8h;
typedef __attribute__((ext_vector_type(8)))  float    v8f;

static __device__ __forceinline__ float leakyf(float x) { return x > 0.f ? x : 0.01f * x; }

// Monotonic float<->uint encoding so integer atomicMax == float max (order independent).
static __device__ __forceinline__ unsigned encf(float x) {
  unsigned u = __float_as_uint(x);
  return (u & 0x80000000u) ? ~u : (u | 0x80000000u);
}
static __device__ __forceinline__ float decf(unsigned u) {
  return __uint_as_float((u & 0x80000000u) ? (u & 0x7fffffffu) : ~u);
}
#define ENC_NEG_INF 0x007fffffu  // encf(-inf)

// Swizzle helpers: map (row/col) of a 128x32 A tile or 32xM B tile into
// fragment-order LDS so each lane reads its whole v16h fragment contiguously.
// A 16x32 f16 fragment layout: lane = h*16+m (h = K-half), elem j -> kk.
static __device__ __forceinline__ int aswz(int r, int c) {
  const int stripe = r >> 4, m = r & 15;
  const int h = (c >> 3) & 1;
  const int v = ((c >> 4) << 2) | ((c & 7) >> 1);
  const int j = (v << 1) | (c & 1);
  return ((stripe * 32 + h * 16 + m) << 4) + j;
}
// B 32x16 f16 fragment layout: lane = ((k>>3)&1)*16 + n, elem j = (k>>4)*8 + (k&7).
static __device__ __forceinline__ int bswz(int k, int n) {
  const int lane = (n & 15) | (((k >> 3) & 1) << 4);
  const int nt = n >> 4;
  const int j = ((k >> 4) << 3) | (k & 7);
  return ((nt * 32 + lane) << 4) + j;
}

// ---------------- utility kernels ----------------
__global__ void k_cvt_f16(const float* __restrict__ s, _Float16* __restrict__ d, long n) {
  long i = (long)blockIdx.x * blockDim.x + threadIdx.x;
  long st = (long)gridDim.x * blockDim.x;
  for (; i < n; i += st) d[i] = (_Float16)s[i];
}
__global__ void k_fill_u32(unsigned* __restrict__ p, long n, unsigned v) {
  long i = (long)blockIdx.x * blockDim.x + threadIdx.x;
  long st = (long)gridDim.x * blockDim.x;
  for (; i < n; i += st) p[i] = v;
}
__global__ void k_zero_f32(float* __restrict__ p, long n) {
  long i = (long)blockIdx.x * blockDim.x + threadIdx.x;
  long st = (long)gridDim.x * blockDim.x;
  for (; i < n; i += st) p[i] = 0.f;
}
__global__ void k_decode_vox(unsigned* __restrict__ p, long n) {
  long i = (long)blockIdx.x * blockDim.x + threadIdx.x;
  long st = (long)gridDim.x * blockDim.x;
  for (; i < n; i += st) {
    unsigned u = p[i];
    ((float*)p)[i] = decf(u);
  }
}

// ------------- deterministic per-feature batch stats (two stage) -------------
template<typename T, int FQ>
__global__ __launch_bounds__(256) void k_stats_partial(const T* __restrict__ X, long Nrows, int F,
                                                       float* __restrict__ partial) {
  __shared__ float ssum[256];
  __shared__ float ssq[256];
  const int t = threadIdx.x;
  const int f = t & (FQ - 1);
  const int rIdx = t / FQ;
  const int RPI = 256 / FQ;
  float sum = 0.f, sq = 0.f;
  if (f < F) {
    for (long r = (long)blockIdx.x * RPI + rIdx; r < Nrows; r += (long)gridDim.x * RPI) {
      float x = (float)X[r * (long)F + f];
      sum += x;
      sq += x * x;
    }
  }
  ssum[t] = sum;
  ssq[t] = sq;
  __syncthreads();
  for (int s = 128; s >= FQ; s >>= 1) {
    if (t < s) { ssum[t] += ssum[t + s]; ssq[t] += ssq[t + s]; }
    __syncthreads();
  }
  if (t < F) {
    partial[(long)blockIdx.x * 2 * F + t] = ssum[t];
    partial[(long)blockIdx.x * 2 * F + F + t] = ssq[t];
  }
}

__global__ void k_stats_final(const float* __restrict__ partial, int nblocks, int F, int KP,
                              long Nrows, const float* __restrict__ g, const float* __restrict__ b,
                              float* __restrict__ st) {
  int f = threadIdx.x;
  if (f >= KP) return;
  float s = 0.f, t = 0.f;
  if (f < F) {
    float sum = 0.f, sq = 0.f;
    for (int i = 0; i < nblocks; ++i) {
      sum += partial[(long)i * 2 * F + f];
      sq  += partial[(long)i * 2 * F + F + f];
    }
    float mean = sum / (float)Nrows;
    float var  = sq / (float)Nrows - mean * mean;  // biased var, matches jnp.var
    float inv  = rsqrtf(var + 1e-5f);
    s = g[f] * inv;
    t = b[f] - mean * s;
  }
  st[f] = s;
  st[KP + f] = t;
}

// ---------------- fused WMMA GEMM ----------------
// Block: 256 threads = 8 waves; each wave owns a 16-row stripe -> 128 rows/block.
// LDS tiles are stored pre-swizzled in fragment order: each lane reads its whole
// v16h fragment with two ds_load_b128 (no scalar u16 gathers).
// EPI: 0 = store f16 activations; 1 = segment-max via encoded atomicMax; 2 = LeakyReLU + grid scatter.
template<typename TIN, int M, int KP, bool AFFINE, bool INRELU, bool VEC, int EPI>
__global__ __launch_bounds__(256) void gemm_wmma(
    const TIN* __restrict__ X, int Nrows, int Ka,
    const float* __restrict__ st,
    const _Float16* __restrict__ Wh,
    const float* __restrict__ bias,
    _Float16* __restrict__ Yout,
    const int* __restrict__ seg, unsigned* __restrict__ voxb,
    const int* __restrict__ unq, float* __restrict__ gout) {
  constexpr int NT = M / 16;
  __shared__ v16h sAs[8 * 32];    // 8 row-stripes x 32 lanes, fragment order (8 KB)
  __shared__ v16h sBs[NT * 32];   // NT col-tiles x 32 lanes, fragment order
  __shared__ float sST[2 * KP];   // folded BN (scale, shift)
  const int lane = threadIdx.x & 31;
  const int wave = threadIdx.x >> 5;
  const int rowBase = blockIdx.x * 128;
  _Float16* sAf = (_Float16*)sAs;
  _Float16* sBf = (_Float16*)sBs;

  if (AFFINE) {
    for (int i = threadIdx.x; i < 2 * KP; i += 256) sST[i] = st[i];
  }

  v8f acc[NT] = {};

  for (int ks = 0; ks < KP; ks += 32) {
    __syncthreads();
    // ---- A tile: fuse producer BN (+ LeakyReLU), narrow to f16, store swizzled ----
    if (VEC) {
      // f16 input, Ka % 32 == 0: 16B vector loads (global_load_b128).
      for (int idx = threadIdx.x; idx < 128 * 4; idx += 256) {
        const int r = idx >> 2, c0 = (idx & 3) << 3;
        const int row = rowBase + r;
        v8h xv = {};
        if (row < Nrows) {
          xv = *(const v8h*)&X[(size_t)row * Ka + ks + c0];
          if (ks + 32 < KP)
            __builtin_prefetch((const void*)&X[(size_t)row * Ka + ks + 32 + c0], 0, 0);
        }
#pragma unroll
        for (int e = 0; e < 8; ++e) {
          const int c = c0 + e;
          float x = (float)xv[e];
          if (AFFINE) x = x * sST[ks + c] + sST[KP + ks + c];
          if (INRELU) x = leakyf(x);
          sAf[aswz(r, c)] = (_Float16)x;
        }
      }
    } else {
      for (int idx = threadIdx.x; idx < 128 * 32; idx += 256) {
        const int r = idx >> 5, c = idx & 31;
        const int row = rowBase + r;
        const int k = ks + c;
        float x = 0.f;
        if (row < Nrows && k < Ka) {
          x = (float)X[(size_t)row * Ka + k];
          if (AFFINE) x = x * sST[k] + sST[KP + k];
          if (INRELU) x = leakyf(x);
        }
        sAf[aswz(r, c)] = (_Float16)x;
      }
    }
    // ---- B tile: f16 weights, 16B vector loads, store swizzled ----
    for (int idx = threadIdx.x; idx < (32 * M) / 8; idx += 256) {
      const int r = idx / (M / 8);
      const int n0 = (idx << 3) & (M - 1);
      const int k = ks + r;
      v8h wv = {};
      if (k < Ka) wv = *(const v8h*)&Wh[(size_t)k * M + n0];
#pragma unroll
      for (int e = 0; e < 8; ++e) sBf[bswz(r, n0 + e)] = wv[e];
    }
    __syncthreads();

    const v16h afrag = sAs[wave * 32 + lane];
#pragma unroll
    for (int nt = 0; nt < NT; ++nt) {
      const v16h bfrag = sBs[nt * 32 + lane];
      acc[nt] = __builtin_amdgcn_wmma_f32_16x16x32_f16(false, afrag, false, bfrag,
                                                       (short)0, acc[nt], false, false);
    }
  }

  // ---- Epilogue. C layout: VGPR r -> M = r + (lane>=16)*8, N = lane&15. ----
  const int nlo = lane & 15;
  const int rowB = rowBase + wave * 16 + ((lane >> 4) << 3);
  float bv[NT];
#pragma unroll
  for (int nt = 0; nt < NT; ++nt) bv[nt] = bias[nt * 16 + nlo];
  const bool full = (rowB + 8 <= Nrows);

  if (EPI == 0) {
    if (full) {
#pragma unroll
      for (int nt = 0; nt < NT; ++nt)
#pragma unroll
        for (int r = 0; r < 8; ++r)
          Yout[(size_t)(rowB + r) * M + nt * 16 + nlo] = (_Float16)(acc[nt][r] + bv[nt]);
    } else {
#pragma unroll
      for (int r = 0; r < 8; ++r)
        if (rowB + r < Nrows)
#pragma unroll
          for (int nt = 0; nt < NT; ++nt)
            Yout[(size_t)(rowB + r) * M + nt * 16 + nlo] = (_Float16)(acc[nt][r] + bv[nt]);
    }
  } else if (EPI == 1) {
    if (full) {
      int sg[8];
#pragma unroll
      for (int r = 0; r < 8; ++r) sg[r] = seg[rowB + r];
#pragma unroll
      for (int nt = 0; nt < NT; ++nt)
#pragma unroll
        for (int r = 0; r < 8; ++r)
          atomicMax(&voxb[(size_t)sg[r] * M + nt * 16 + nlo], encf(acc[nt][r] + bv[nt]));
    } else {
#pragma unroll
      for (int r = 0; r < 8; ++r)
        if (rowB + r < Nrows) {
          const int sg = seg[rowB + r];
#pragma unroll
          for (int nt = 0; nt < NT; ++nt)
            atomicMax(&voxb[(size_t)sg * M + nt * 16 + nlo], encf(acc[nt][r] + bv[nt]));
        }
    }
  } else {
    if (full) {
      int bi[8], hi[8], wi[8];
#pragma unroll
      for (int r = 0; r < 8; ++r) {
        bi[r] = unq[(rowB + r) * 3 + 0];
        hi[r] = unq[(rowB + r) * 3 + 1];
        wi[r] = unq[(rowB + r) * 3 + 2];
      }
#pragma unroll
      for (int nt = 0; nt < NT; ++nt)
#pragma unroll
        for (int r = 0; r < 8; ++r) {
          const float v = leakyf(acc[nt][r] + bv[nt]);
          gout[(((size_t)bi[r] * 32 + nt * 16 + nlo) * 480 + hi[r]) * 360 + wi[r]] = v;
        }
    } else {
#pragma unroll
      for (int r = 0; r < 8; ++r)
        if (rowB + r < Nrows) {
          const int bb = unq[(rowB + r) * 3 + 0];
          const int hh = unq[(rowB + r) * 3 + 1];
          const int ww = unq[(rowB + r) * 3 + 2];
#pragma unroll
          for (int nt = 0; nt < NT; ++nt) {
            const float v = leakyf(acc[nt][r] + bv[nt]);
            gout[(((size_t)bb * 32 + nt * 16 + nlo) * 480 + hh) * 360 + ww] = v;
          }
        }
    }
  }
}

// ---------------- host orchestration ----------------
extern "C" void kernel_launch(void* const* d_in, const int* in_sizes, int n_in,
                              void* d_out, int out_size, void* d_ws, size_t ws_size,
                              hipStream_t stream) {
  const float* fea     = (const float*)d_in[0];
  const int*   unq     = (const int*)d_in[1];
  const int*   unq_inv = (const int*)d_in[2];
  // d_in[3] = n (batch, == 2, baked into the output layout)
  const float* g0 = (const float*)d_in[4],  *b0 = (const float*)d_in[5];
  const float* W1 = (const float*)d_in[6],  *bl1 = (const float*)d_in[7];
  const float* g1 = (const float*)d_in[8],  *b1 = (const float*)d_in[9];
  const float* W2 = (const float*)d_in[10], *bl2 = (const float*)d_in[11];
  const float* g2 = (const float*)d_in[12], *b2 = (const float*)d_in[13];
  const float* W3 = (const float*)d_in[14], *bl3 = (const float*)d_in[15];
  const float* g3 = (const float*)d_in[16], *b3 = (const float*)d_in[17];
  const float* W4 = (const float*)d_in[18], *bl4 = (const float*)d_in[19];
  const float* Wf = (const float*)d_in[20], *blf = (const float*)d_in[21];

  const long N  = in_sizes[0] / 9;   // 400000 points
  const long MV = in_sizes[1] / 3;   // 120000 voxels

  // ---- workspace layout (y1+y2 region reused for vox buffer once dead) ----
  char* ws = (char*)d_ws;
  size_t off = 0;
  auto alloc = [&](size_t bytes) {
    size_t o = off;
    off += (bytes + 255) & ~(size_t)255;
    return o;
  };
  _Float16* W1h = (_Float16*)(ws + alloc(9 * 64 * 2));
  _Float16* W2h = (_Float16*)(ws + alloc(64 * 128 * 2));
  _Float16* W3h = (_Float16*)(ws + alloc(128 * 256 * 2));
  _Float16* W4h = (_Float16*)(ws + alloc(256 * 256 * 2));
  _Float16* Wfh = (_Float16*)(ws + alloc(256 * 32 * 2));
  float* st0 = (float*)(ws + alloc(2 * 32 * 4));
  float* st1 = (float*)(ws + alloc(2 * 64 * 4));
  float* st2 = (float*)(ws + alloc(2 * 128 * 4));
  float* st3 = (float*)(ws + alloc(2 * 256 * 4));
  const int SBLK = 256;  // stats stage-1 blocks (fixed => deterministic)
  float* partial = (float*)(ws + alloc((size_t)SBLK * 2 * 256 * 4));
  _Float16* y3 = (_Float16*)(ws + alloc((size_t)N * 256 * 2));
  size_t y1off = alloc((size_t)N * 64 * 2);
  size_t y2off = alloc((size_t)N * 128 * 2);
  _Float16* y1 = (_Float16*)(ws + y1off);
  _Float16* y2 = (_Float16*)(ws + y2off);
  unsigned* vox = (unsigned*)(ws + y1off);  // 120000*256*4 = 123 MB <= |y1|+|y2| = 154 MB
  (void)ws_size; (void)n_in; (void)y2off;

  const dim3 B256(256);
  const dim3 GU(2048);
  const dim3 GN((unsigned)((N + 127) / 128));
  const dim3 GV((unsigned)((MV + 127) / 128));

  // Weights fp32 -> f16.
  k_cvt_f16<<<GU, B256, 0, stream>>>(W1, W1h, 9 * 64);
  k_cvt_f16<<<GU, B256, 0, stream>>>(W2, W2h, 64 * 128);
  k_cvt_f16<<<GU, B256, 0, stream>>>(W3, W3h, 128 * 256);
  k_cvt_f16<<<GU, B256, 0, stream>>>(W4, W4h, 256 * 256);
  k_cvt_f16<<<GU, B256, 0, stream>>>(Wf, Wfh, 256 * 32);

  // BN0 stats on fea (9 features, padded to K=32).
  k_stats_partial<float, 16><<<SBLK, B256, 0, stream>>>(fea, N, 9, partial);
  k_stats_final<<<1, 32, 0, stream>>>(partial, SBLK, 9, 32, N, g0, b0, st0);

  // Layer 1: BN0(fea) @ W1 + bl1 -> y1 [N,64]
  gemm_wmma<float, 64, 32, true, false, false, 0><<<GN, B256, 0, stream>>>(
      fea, (int)N, 9, st0, W1h, bl1, y1, nullptr, nullptr, nullptr, nullptr);
  k_stats_partial<_Float16, 64><<<SBLK, B256, 0, stream>>>(y1, N, 64, partial);
  k_stats_final<<<1, 64, 0, stream>>>(partial, SBLK, 64, 64, N, g1, b1, st1);

  // Layer 2: lrelu(BN1(y1)) @ W2 + bl2 -> y2 [N,128]
  gemm_wmma<_Float16, 128, 64, true, true, true, 0><<<GN, B256, 0, stream>>>(
      y1, (int)N, 64, st1, W2h, bl2, y2, nullptr, nullptr, nullptr, nullptr);
  k_stats_partial<_Float16, 128><<<SBLK, B256, 0, stream>>>(y2, N, 128, partial);
  k_stats_final<<<1, 128, 0, stream>>>(partial, SBLK, 128, 128, N, g2, b2, st2);

  // Layer 3: lrelu(BN2(y2)) @ W3 + bl3 -> y3 [N,256]
  gemm_wmma<_Float16, 256, 128, true, true, true, 0><<<GN, B256, 0, stream>>>(
      y2, (int)N, 128, st2, W3h, bl3, y3, nullptr, nullptr, nullptr, nullptr);
  k_stats_partial<_Float16, 256><<<SBLK, B256, 0, stream>>>(y3, N, 256, partial);
  k_stats_final<<<1, 256, 0, stream>>>(partial, SBLK, 256, 256, N, g3, b3, st3);

  // Layer 4 fused with segment_max: lrelu(BN3(y3)) @ W4 + bl4, atomicMax into vox.
  k_fill_u32<<<GU, B256, 0, stream>>>(vox, MV * 256, ENC_NEG_INF);
  gemm_wmma<_Float16, 256, 256, true, true, true, 1><<<GN, B256, 0, stream>>>(
      y3, (int)N, 256, st3, W4h, bl4, nullptr, unq_inv, vox, nullptr, nullptr);

  // Decode encoded maxima back to f32 in place.
  k_decode_vox<<<GU, B256, 0, stream>>>(vox, MV * 256);

  // Zero the dense output grid (harness poisons it), then final GEMM + scatter.
  k_zero_f32<<<GU, B256, 0, stream>>>((float*)d_out, (long)out_size);
  gemm_wmma<float, 32, 256, false, false, false, 2><<<GV, B256, 0, stream>>>(
      (const float*)vox, (int)MV, 256, nullptr, Wfh, blf, nullptr, nullptr, nullptr,
      unq, (float*)d_out);
}
